// block_23407571763490
// MI455X (gfx1250) — matrix operand order repaired and checked by reading, not compiled, not run on hardware
//
#include <hip/hip_runtime.h>
#include <hip/hip_bf16.h>
#include <math.h>

// ---------------------------------------------------------------------------
// CDNA5 (gfx1250) transformer block forward.
// bf16 WMMA (V_WMMA_F32_16X16X32_BF16, f32 accumulate) for all GEMMs.
// B-operand tiles (128x64 bf16, 16KB) staged to LDS by the Tensor Data Mover
// (tensor_load_to_lds, TENSORcnt pipelined, double-buffered);
// fp32 norms / softmax / residuals.
// ---------------------------------------------------------------------------

typedef __attribute__((ext_vector_type(16))) __bf16 v16bf;
typedef __attribute__((ext_vector_type(8)))  __bf16 v8bf;
typedef __attribute__((ext_vector_type(8)))  float  v8f;
typedef __attribute__((ext_vector_type(4)))  unsigned int u32x4;
typedef __attribute__((ext_vector_type(8)))  int i32x8;
typedef __attribute__((ext_vector_type(4)))  int i32x4;

static constexpr int Bn = 2;
static constexpr int Tn = 2048;
static constexpr int Dn = 1024;
static constexpr int Mn = Bn * Tn;   // 4096 rows when batch is folded

// ------------------------------- fp32 -> bf16 ------------------------------
__global__ void k_cvt_bf16(const float* __restrict__ in, __bf16* __restrict__ out, int n) {
    int i = blockIdx.x * blockDim.x + threadIdx.x;
    if (i < n) out[i] = (__bf16)in[i];
}

// ------------------------------- RMSNorm -----------------------------------
// y = x / (sqrt(mean(x^2)) + 1e-6) * w      (eps OUTSIDE sqrt, per reference)
__global__ __launch_bounds__(256) void k_rmsnorm(const float* __restrict__ x,
                                                 const float* __restrict__ w,
                                                 __bf16* __restrict__ out, int D) {
    __shared__ float red[256];
    const int row = blockIdx.x;
    const float* xr = x + (size_t)row * D;
    float s = 0.f;
    for (int j = threadIdx.x; j < D; j += 256) { float v = xr[j]; s += v * v; }
    red[threadIdx.x] = s;
    __syncthreads();
    for (int st = 128; st > 0; st >>= 1) {
        if (threadIdx.x < st) red[threadIdx.x] += red[threadIdx.x + st];
        __syncthreads();
    }
    const float inv = 1.f / (sqrtf(red[0] / (float)D) + 1e-6f);
    __bf16* orow = out + (size_t)row * D;
    for (int j = threadIdx.x; j < D; j += 256) orow[j] = (__bf16)(xr[j] * inv * w[j]);
}

// --------------------------- causal softmax row ----------------------------
// val = (k <= q) ? S*scale : -1e9 ;  P = softmax(val)  -> bf16
__global__ __launch_bounds__(256) void k_softmax_causal(const float* __restrict__ S,
                                                        __bf16* __restrict__ P,
                                                        int T, float scale) {
    __shared__ float red[256];
    const int q = blockIdx.x;
    const int b = blockIdx.y;
    const float* row = S + ((size_t)b * T + q) * T;
    __bf16* prow = P + ((size_t)b * T + q) * T;
    const int tid = threadIdx.x;

    float mx = -1e30f;
    for (int j = tid; j < T; j += 256) {
        float v = (j <= q) ? row[j] * scale : -1e9f;
        mx = fmaxf(mx, v);
    }
    red[tid] = mx; __syncthreads();
    for (int st = 128; st > 0; st >>= 1) {
        if (tid < st) red[tid] = fmaxf(red[tid], red[tid + st]);
        __syncthreads();
    }
    mx = red[0];
    __syncthreads();

    float sum = 0.f;
    for (int j = tid; j < T; j += 256) {
        float v = (j <= q) ? row[j] * scale : -1e9f;
        sum += __expf(v - mx);
    }
    red[tid] = sum; __syncthreads();
    for (int st = 128; st > 0; st >>= 1) {
        if (tid < st) red[tid] += red[tid + st];
        __syncthreads();
    }
    const float inv = 1.f / red[0];
    for (int j = tid; j < T; j += 256) {
        float v = (j <= q) ? row[j] * scale : -1e9f;
        prow[j] = (__bf16)(__expf(v - mx) * inv);
    }
}

// -------------------- TDM: 128(N) x 64(K) bf16 tile -> LDS -----------------
// D# per CDNA5 ISA. 2-D tensor W[N,K] row-major,
// tile start at (row n0, col k). Groups 2/3 zero (<=2D). Tracked by TENSORcnt.
__device__ static inline void tdm_load_btile(const __bf16* gsrc, unsigned lds_off,
                                             int K, int N) {
    unsigned long long ga = (unsigned long long)(uintptr_t)gsrc;
    u32x4 g0;
    g0[0] = 1u;                                          // count=1, flags=0
    g0[1] = lds_off;                                     // lds_addr (bytes)
    g0[2] = (unsigned)(ga & 0xFFFFFFFFu);                // global_addr[31:0]
    g0[3] = (unsigned)((ga >> 32) & 0x1FFFFFFu)          // global_addr[56:32]
            | (2u << 30);                                // type=2 ("image")
    i32x8 g1;
    g1[0] = 0x00010000;                                  // data_size=1 (2 bytes)
    g1[1] = (K & 0xFFFF) << 16;                          // tensor_dim0[15:0]
    g1[2] = ((K >> 16) & 0xFFFF) | ((N & 0xFFFF) << 16); // dim0 hi | dim1 lo
    g1[3] = ((N >> 16) & 0xFFFF) | (64 << 16);           // dim1 hi | tile_dim0=64
    g1[4] = 128;                                         // tile_dim1=128, tile_dim2=0
    g1[5] = K;                                           // tensor_dim0_stride[31:0]
    g1[6] = 0;                                           // stride hi, dim1_stride lo
    g1[7] = 0;
    i32x4 z4 = {0, 0, 0, 0};
    i32x8 z8 = {0, 0, 0, 0, 0, 0, 0, 0};
    __builtin_amdgcn_tensor_load_to_lds(g0, g1, z4, z4, z8, 0);
}

// ------------------------------ WMMA GEMM ----------------------------------
// C[M,N] = A[M,K](bf16) @ W[N,K](bf16)^T  (+bias) (+GELU) (+resid), batched
// via grid.z with element strides aBatch/wBatch/oBatch.
// Block = 8 waves = 128 rows x 128 cols; each wave a 16x128 strip.
// B tile (128 rows x 64 K, shared by all 8 waves) is TDM-loaded into
// double-buffered LDS; 16 WMMAs per wave per barrier interval.
template <bool OUT_BF16, bool GELU, bool HAS_BIAS, bool HAS_RESID,
          bool TRANS_OUT, bool CAUSAL>
__global__ __launch_bounds__(256)
void gemm_wmma(const __bf16* __restrict__ A, const __bf16* __restrict__ W,
               const float* __restrict__ bias, const float* __restrict__ resid,
               void* __restrict__ out, int M, int N, int K,
               unsigned long long aBatch, unsigned long long wBatch,
               unsigned long long oBatch) {
    __shared__ __align__(16) __bf16 bs[2][128 * 64];     // 2 x 16KB

    const int wave = threadIdx.x >> 5;
    const int lane = threadIdx.x & 31;
    const int kseg = lane >> 4;   // which 16-lane half of the wave
    const int lrow = lane & 15;

    const int m0 = blockIdx.y * 128 + wave * 16;
    const int n0 = blockIdx.x * 128;
    // Block-uniform causal skip (whole 128-row block above the diagonal);
    // must be uniform because of the barriers in the K-loop.
    if (CAUSAL && n0 > (int)blockIdx.y * 128 + 127) return;

    const __bf16* Ab = A + (size_t)blockIdx.z * aBatch;
    const __bf16* Wb = W + (size_t)blockIdx.z * wBatch;

    // A-frag: lane holds row (m0+lrow); 8 bf16 at K+kseg*8 and 8 at K+16+kseg*8
    const __bf16* arow = Ab + (size_t)(m0 + lrow) * K + kseg * 8;
    // B tile base rows in global: W rows n0..n0+127
    const __bf16* wtile = Wb + (size_t)n0 * K;

    const unsigned lds0 = (unsigned)(uintptr_t)(void*)&bs[0][0];
    const unsigned lds1 = (unsigned)(uintptr_t)(void*)&bs[1][0];
    const int iters = K / 64;

    if (wave == 0) tdm_load_btile(wtile, lds0, K, N);     // prologue tile 0

    v8f acc[8] = {};
    for (int i = 0; i < iters; ++i) {
        const int k = i * 64;
        if (wave == 0) {
            if (i + 1 < iters) {
                tdm_load_btile(wtile + (k + 64), (i & 1) ? lds0 : lds1, K, N);
                __builtin_amdgcn_s_wait_tensorcnt(1);     // tile i landed
            } else {
                __builtin_amdgcn_s_wait_tensorcnt(0);     // tail
            }
        }
        __syncthreads();                                  // tile i visible to all

        // two 16x32 A fragments covering K = k..k+63
        v8bf a0lo = *(const v8bf*)(arow + k);
        v8bf a0hi = *(const v8bf*)(arow + k + 16);
        v8bf a1lo = *(const v8bf*)(arow + k + 32);
        v8bf a1hi = *(const v8bf*)(arow + k + 48);
        v16bf a0 = __builtin_shufflevector(a0lo, a0hi,
                    0, 1, 2, 3, 4, 5, 6, 7, 8, 9, 10, 11, 12, 13, 14, 15);
        v16bf a1 = __builtin_shufflevector(a1lo, a1hi,
                    0, 1, 2, 3, 4, 5, 6, 7, 8, 9, 10, 11, 12, 13, 14, 15);

        const __bf16* bb = &bs[i & 1][0];   // LDS tile: [128 rows][64 K] bf16
#pragma unroll
        for (int t = 0; t < 8; ++t) {
            // lane's B-frags: LDS row (t*16+lrow), 8 bf16 at K+kseg*8 and 8 at K+16+kseg*8
            const __bf16* brow = bb + (t * 16 + lrow) * 64 + kseg * 8;
            v16bf b0 = __builtin_shufflevector(*(const v8bf*)(brow), *(const v8bf*)(brow + 16),
                    0, 1, 2, 3, 4, 5, 6, 7, 8, 9, 10, 11, 12, 13, 14, 15);
            v16bf b1 = __builtin_shufflevector(*(const v8bf*)(brow + 32), *(const v8bf*)(brow + 48),
                    0, 1, 2, 3, 4, 5, 6, 7, 8, 9, 10, 11, 12, 13, 14, 15);
            acc[t] = __builtin_amdgcn_wmma_f32_16x16x32_bf16(
                false, a0, false, b0, (short)0, acc[t], false, false);
            acc[t] = __builtin_amdgcn_wmma_f32_16x16x32_bf16(
                false, a1, false, b1, (short)0, acc[t], false, false);
        }
        __syncthreads();                                  // done reading buf i&1
    }

    // Epilogue. C layout: reg r of lane -> row m0 + r + 8*kseg, col n0sub + lrow.
#pragma unroll
    for (int t = 0; t < 8; ++t) {
#pragma unroll
        for (int r = 0; r < 8; ++r) {
            const int rowg = m0 + r + 8 * kseg;
            const int colg = n0 + t * 16 + lrow;
            float v = acc[t][r];
            if (HAS_BIAS) v += bias[colg];
            if (GELU)     v = 0.5f * v * (1.f + erff(v * 0.70710678118654752f));
            if (HAS_RESID) v += resid[(size_t)rowg * N + colg];
            if (TRANS_OUT) {
                __bf16* ob = (__bf16*)out + (size_t)blockIdx.z * oBatch;
                ob[(size_t)colg * M + rowg] = (__bf16)v;   // out[N,M] (V^T)
            } else if (OUT_BF16) {
                __bf16* ob = (__bf16*)out + (size_t)blockIdx.z * oBatch;
                ob[(size_t)rowg * N + colg] = (__bf16)v;
            } else {
                float* of = (float*)out + (size_t)blockIdx.z * oBatch;
                of[(size_t)rowg * N + colg] = v;
            }
        }
    }
}

// ------------------------------- launcher ----------------------------------
extern "C" void kernel_launch(void* const* d_in, const int* in_sizes, int n_in,
                              void* d_out, int out_size, void* d_ws, size_t ws_size,
                              hipStream_t stream) {
    (void)in_sizes; (void)n_in; (void)out_size; (void)ws_size;

    const float* x    = (const float*)d_in[0];
    const float* n1_w = (const float*)d_in[1];
    const float* n2_w = (const float*)d_in[2];
    const float* Wq   = (const float*)d_in[3];
    const float* bq   = (const float*)d_in[4];
    const float* Wk   = (const float*)d_in[5];
    const float* bk   = (const float*)d_in[6];
    const float* Wv   = (const float*)d_in[7];
    const float* bv   = (const float*)d_in[8];
    const float* Wo   = (const float*)d_in[9];
    const float* bo   = (const float*)d_in[10];
    const float* W1   = (const float*)d_in[11];
    const float* b1   = (const float*)d_in[12];
    const float* W2   = (const float*)d_in[13];
    const float* b2   = (const float*)d_in[14];

    // ----- workspace layout (bytes; ~120 MB total) -----
    const size_t MB = 1ull << 20;
    char* ws = (char*)d_ws;
    __bf16* wq_b = (__bf16*)(ws + 0 * MB);    // D*D bf16   = 2MB
    __bf16* wk_b = (__bf16*)(ws + 2 * MB);
    __bf16* wv_b = (__bf16*)(ws + 4 * MB);
    __bf16* wo_b = (__bf16*)(ws + 6 * MB);
    __bf16* w1_b = (__bf16*)(ws + 8 * MB);    // 4D*D bf16  = 8MB
    __bf16* w2_b = (__bf16*)(ws + 16 * MB);   // 8MB
    __bf16* h_b  = (__bf16*)(ws + 24 * MB);   // [4096,1024] bf16 = 8MB (norm1, then norm2)
    __bf16* q_b  = (__bf16*)(ws + 32 * MB);   // 8MB (reused as attn_out bf16)
    __bf16* k_b  = (__bf16*)(ws + 40 * MB);   // 8MB
    __bf16* vt_b = (__bf16*)(ws + 48 * MB);   // [B][D][T] bf16 = 8MB
    float*  Sbuf = (float*) (ws + 56 * MB);   // [B][T][T] f32 = 32MB (reused as MLP mid bf16)
    __bf16* Pbuf = (__bf16*)(ws + 88 * MB);   // [B][T][T] bf16 = 16MB
    float*  x1   = (float*) (ws + 104 * MB);  // [4096,1024] f32 = 16MB
    __bf16* ao_b  = q_b;                      // alias: q dead after S
    __bf16* mid_b = (__bf16*)Sbuf;            // alias: S dead after softmax

    // 1) weights fp32 -> bf16
    k_cvt_bf16<<<(Dn * Dn + 255) / 256, 256, 0, stream>>>(Wq, wq_b, Dn * Dn);
    k_cvt_bf16<<<(Dn * Dn + 255) / 256, 256, 0, stream>>>(Wk, wk_b, Dn * Dn);
    k_cvt_bf16<<<(Dn * Dn + 255) / 256, 256, 0, stream>>>(Wv, wv_b, Dn * Dn);
    k_cvt_bf16<<<(Dn * Dn + 255) / 256, 256, 0, stream>>>(Wo, wo_b, Dn * Dn);
    k_cvt_bf16<<<(4 * Dn * Dn + 255) / 256, 256, 0, stream>>>(W1, w1_b, 4 * Dn * Dn);
    k_cvt_bf16<<<(4 * Dn * Dn + 255) / 256, 256, 0, stream>>>(W2, w2_b, 4 * Dn * Dn);

    // 2) h = rmsnorm(x, n1_w)
    k_rmsnorm<<<Mn, 256, 0, stream>>>(x, n1_w, h_b, Dn);

    // 3/4) q = h@Wq^T+bq, k = h@Wk^T+bk   (bf16 out)
    gemm_wmma<true, false, true, false, false, false>
        <<<dim3(Dn / 128, Mn / 128, 1), 256, 0, stream>>>(
            h_b, wq_b, bq, nullptr, q_b, Mn, Dn, Dn, 0, 0, 0);
    gemm_wmma<true, false, true, false, false, false>
        <<<dim3(Dn / 128, Mn / 128, 1), 256, 0, stream>>>(
            h_b, wk_b, bk, nullptr, k_b, Mn, Dn, Dn, 0, 0, 0);

    // 5) v^T = (h@Wv^T+bv)^T per batch -> vt[b][D][T]  (transposed bf16 store)
    gemm_wmma<true, false, true, false, true, false>
        <<<dim3(Dn / 128, Tn / 128, Bn), 256, 0, stream>>>(
            h_b, wv_b, bv, nullptr, vt_b, Tn, Dn, Dn,
            (unsigned long long)Tn * Dn, 0, (unsigned long long)Dn * Tn);

    // 6) S = q @ k^T per batch (f32, causal blocks above diagonal skipped)
    gemm_wmma<false, false, false, false, false, true>
        <<<dim3(Tn / 128, Tn / 128, Bn), 256, 0, stream>>>(
            q_b, k_b, nullptr, nullptr, Sbuf, Tn, Tn, Dn,
            (unsigned long long)Tn * Dn, (unsigned long long)Tn * Dn,
            (unsigned long long)Tn * Tn);

    // 7) P = softmax(mask(S/32))  -> bf16
    k_softmax_causal<<<dim3(Tn, Bn), 256, 0, stream>>>(Sbuf, Pbuf, Tn, 1.f / 32.f);

    // 8) attn_out = P @ v  (= P @ (v^T)^T) per batch  -> bf16
    gemm_wmma<true, false, false, false, false, false>
        <<<dim3(Dn / 128, Tn / 128, Bn), 256, 0, stream>>>(
            Pbuf, vt_b, nullptr, nullptr, ao_b, Tn, Dn, Tn,
            (unsigned long long)Tn * Tn, (unsigned long long)Dn * Tn,
            (unsigned long long)Tn * Dn);

    // 9) x1 = x + attn_out @ Wo^T + bo  (f32)
    gemm_wmma<false, false, true, true, false, false>
        <<<dim3(Dn / 128, Mn / 128, 1), 256, 0, stream>>>(
            ao_b, wo_b, bo, x, x1, Mn, Dn, Dn, 0, 0, 0);

    // 10) h = rmsnorm(x1, n2_w)
    k_rmsnorm<<<Mn, 256, 0, stream>>>(x1, n2_w, h_b, Dn);

    // 11) mid = gelu(h @ W1^T + b1)  (bf16)
    gemm_wmma<true, true, true, false, false, false>
        <<<dim3(4 * Dn / 128, Mn / 128, 1), 256, 0, stream>>>(
            h_b, w1_b, b1, nullptr, mid_b, Mn, 4 * Dn, Dn, 0, 0, 0);

    // 12) out = x1 + mid @ W2^T + b2  (f32)
    gemm_wmma<false, false, true, true, false, false>
        <<<dim3(Dn / 128, Mn / 128, 1), 256, 0, stream>>>(
            mid_b, w2_b, b2, x1, d_out, Mn, Dn, 4 * Dn, 0, 0, 0);
}
